// GlobalCrossAttention_21045339750660
// MI455X (gfx1250) — matrix operand order repaired and checked by reading, not compiled, not run on hardware
//
#include <hip/hip_runtime.h>

typedef __attribute__((ext_vector_type(16))) _Float16 v16h;
typedef __attribute__((ext_vector_type(8)))  float    v8f;
typedef __attribute__((ext_vector_type(2)))  float    v2f;
typedef __attribute__((ext_vector_type(4)))  unsigned int v4u;

namespace {
constexpr int   kB     = 2;
constexpr int   kNQ    = 900;
constexpr int   kNK    = 10000;
constexpr int   kC     = 256;
constexpr int   kH     = 8;
constexpr int   kD     = 32;
constexpr int   kNQP   = 912;                  // 57*16 padded query rows
constexpr int   kSplit = 4;                    // waves per block (split-K)
constexpr int   kFullChunks = kNK / 32;        // 312 full 32-key chunks
constexpr int   kTail  = kFullChunks * 32;     // 9984, 16-key tail
constexpr float kScale = 0.17677669529663687f; // 1/sqrt(32)
}

// ---- DPP 16-lane butterfly reduction (stays within each 16-lane half) ------
template <int CTRL>
__device__ __forceinline__ float dpp_mov(float x) {
    return __builtin_bit_cast(float,
        __builtin_amdgcn_update_dpp(0, __builtin_bit_cast(int, x),
                                    CTRL, 0xf, 0xf, true));
}
__device__ __forceinline__ float red_max16(float x) {
    x = fmaxf(x, dpp_mov<0xB1>(x));   // quad_perm(1,0,3,2)  : xor 1
    x = fmaxf(x, dpp_mov<0x4E>(x));   // quad_perm(2,3,0,1)  : xor 2
    x = fmaxf(x, dpp_mov<0x141>(x));  // row_half_mirror     : combine quads
    x = fmaxf(x, dpp_mov<0x140>(x));  // row_mirror          : combine halves
    return x;
}

// ---------------------------------------------------------------------------
// Generic 16x16-tile projection GEMM:  Y = X[M,256] @ W[256,256] + bias (+pos)
// MODE 0: f16 store, head layout [b][h][n][d] (skip OOB rows)
// MODE 1: f16 store, transposed  [b][h][d][NP]
// MODE 2: f32 store, row major   [M][256]
// ---------------------------------------------------------------------------
template <int MODE>
__global__ __launch_bounds__(32) void proj_kernel(
    const float* __restrict__ X, const float* __restrict__ W,
    const float* __restrict__ bias, const float* __restrict__ pos,
    int Mreal, int rowsPerBatch, int NP, float scale,
    _Float16* __restrict__ outH, float* __restrict__ outF)
{
    const int lane   = threadIdx.x;
    const int lane15 = lane & 15;
    const int hi     = lane >> 4;
    const int m0     = blockIdx.x * 16;
    const int n0     = blockIdx.y * 16;
    const int coln   = n0 + lane15;

    v8f c = {};

#if __has_builtin(__builtin_amdgcn_wmma_f32_16x16x4_f32)
    int rowA = m0 + lane15; if (rowA >= Mreal) rowA = Mreal - 1;
    const float* xrow = X + (size_t)rowA * kC;
    #pragma unroll 4
    for (int k0 = 0; k0 < kC; k0 += 4) {
        v2f a, b;
        a.x = xrow[k0 + hi * 2];
        a.y = xrow[k0 + hi * 2 + 1];
        b.x = W[(size_t)(k0 + hi * 2)     * kC + coln];
        b.y = W[(size_t)(k0 + hi * 2 + 1) * kC + coln];
        c = __builtin_amdgcn_wmma_f32_16x16x4_f32(false, a, false, b,
                                                  (short)0, c, false, false);
    }
#else
    #pragma unroll
    for (int r = 0; r < 8; ++r) {
        int row = m0 + hi * 8 + r; if (row >= Mreal) row = Mreal - 1;
        float acc = 0.f;
        for (int k = 0; k < kC; ++k)
            acc += X[(size_t)row * kC + k] * W[(size_t)k * kC + coln];
        c[r] = acc;
    }
#endif

    const float bcol = bias[coln];
    #pragma unroll
    for (int r = 0; r < 8; ++r) {
        const int  row   = m0 + hi * 8 + r;
        const bool valid = (row < Mreal);
        const int  rowc  = valid ? row : (Mreal - 1);
        float y = c[r] + bcol;
        if (pos) y += pos[(size_t)rowc * kC + coln];
        y *= scale;
        const int bb = rowc / rowsPerBatch;
        const int n  = rowc % rowsPerBatch;
        const int h  = coln >> 5;
        const int d  = coln & 31;
        if (MODE == 0) {
            if (valid)
                outH[(((size_t)bb * kH + h) * NP + n) * kD + d] = (_Float16)y;
        } else if (MODE == 1) {
            if (valid)
                outH[(((size_t)bb * kH + h) * kD + d) * (size_t)NP + n] = (_Float16)y;
        } else {
            if (valid)
                outF[(size_t)row * kC + coln] = y;
        }
    }
}

// ---------------------------------------------------------------------------
// Split-K flash attention: 4 waves per (b, h, 16-query tile); wave w owns key
// chunks w, w+4, w+8, ... (32 keys each). Exact LSE merge through LDS.
// ---------------------------------------------------------------------------
__global__ __launch_bounds__(32 * kSplit) void attn_kernel(
    const _Float16* __restrict__ Qh,   // [B][H][NQP][D] f16, pre-scaled
    const _Float16* __restrict__ Kh,   // [B][H][NK][D]  f16
    const _Float16* __restrict__ Vt,   // [B][H][D][NK]  f16 (transposed)
    const float*    __restrict__ rpe,  // [B][H][NQ][NK] f32
    const int*      __restrict__ mask, // [B][NK] int32
    float*          __restrict__ Xout) // [B][NQ][C] f32
{
    __shared__ __attribute__((aligned(32))) _Float16 Plds[kSplit][16 * 32];
    __shared__ float Olds[kSplit][16][32];
    __shared__ float Mlds[kSplit][16];
    __shared__ float Llds[kSplit][16];

    const int tid    = threadIdx.x;
    const int w      = tid >> 5;       // wave id (split index), uniform in wave
    const int lane   = tid & 31;
    const int lane15 = lane & 15;
    const int hi     = lane >> 4;
    const int q0     = blockIdx.x * 16;
    const int h      = blockIdx.y;
    const int b      = blockIdx.z;
    const int bh     = b * kH + h;

    // Q A-operand (loop invariant). lane<16: M=lane, K=0..15;
    // lane>=16: M=lane-16, K=16..31.
    union { v16h v; v4u u[2]; } aq;
    {
        const _Float16* qrow = Qh + (((size_t)bh) * kNQP + q0 + lane15) * kD;
        aq.u[0] = *(const v4u*)(qrow + hi * 16);
        aq.u[1] = *(const v4u*)(qrow + hi * 16 + 8);
    }

    const float* rpr[8];
    #pragma unroll
    for (int r = 0; r < 8; ++r) {
        int qq = q0 + hi * 8 + r; if (qq >= kNQ) qq = kNQ - 1;
        rpr[r] = rpe + ((size_t)bh * kNQ + qq) * kNK;
    }

    const _Float16* kbase = Kh + (size_t)bh * kNK * kD;
    const _Float16* vbase = Vt + (size_t)bh * kD * kNK;
    const int*      mrow  = mask + (size_t)b * kNK;

    v16h bones;
    #pragma unroll
    for (int i = 0; i < 16; ++i) bones[i] = (_Float16)1.0f;

    float m_[8], l_[8];
    v8f o0 = {}, o1 = {};
    #pragma unroll
    for (int r = 0; r < 8; ++r) { m_[r] = -__builtin_inff(); l_[r] = 0.f; }

    auto chunk = [&](int kk0, bool full) {
        // ---- S = Q*K^T
        union { v16h v; v4u u[2]; } bk;
        v8f s0 = {}, s1 = {};
        {
            const _Float16* kp = kbase + (size_t)(kk0 + lane15) * kD + hi * 16;
            bk.u[0] = *(const v4u*)(kp);
            bk.u[1] = *(const v4u*)(kp + 8);
        }
        s0 = __builtin_amdgcn_wmma_f32_16x16x32_f16(false, aq.v, false, bk.v,
                                                    (short)0, s0, false, false);
        if (full) {
            const _Float16* kp = kbase + (size_t)(kk0 + 16 + lane15) * kD + hi * 16;
            bk.u[0] = *(const v4u*)(kp);
            bk.u[1] = *(const v4u*)(kp + 8);
            s1 = __builtin_amdgcn_wmma_f32_16x16x32_f16(false, aq.v, false, bk.v,
                                                        (short)0, s1, false, false);
        }

        // ---- rpe (non-temporal stream) + soft padding mask; prefetch next
        const float mk0 = -100.f * (float)mrow[kk0 + lane15];
        const float mk1 = full ? -100.f * (float)mrow[kk0 + 16 + lane15] : 0.f;
        #pragma unroll
        for (int r = 0; r < 8; ++r) {
            const float* rp = rpr[r] + kk0 + lane15;
            s0[r] += __builtin_nontemporal_load(rp) + mk0;
            if (full) s1[r] += __builtin_nontemporal_load(rp + 16) + mk1;
        }
        if (full) {
            #pragma unroll
            for (int r = 0; r < 8; ++r)
                __builtin_prefetch(rpr[r] + kk0 + lane15 + 32 * kSplit, 0, 0);
        }

        // ---- online softmax: DPP max-reduce per row, sum via ones-WMMA
        float p0[8], p1[8], al[8];
        #pragma unroll
        for (int r = 0; r < 8; ++r) {
            float t = full ? fmaxf(s0[r], s1[r]) : s0[r];
            t = red_max16(t);
            const float nm = fmaxf(m_[r], t);
            al[r] = __expf(m_[r] - nm);
            m_[r] = nm;
            p0[r] = __expf(s0[r] - nm);
            p1[r] = full ? __expf(s1[r] - nm) : 0.f;
            o0[r] *= al[r];
            o1[r] *= al[r];
        }

        // ---- P: C-layout -> A-layout through this wave's LDS slice
        __builtin_amdgcn_wave_barrier();
        #pragma unroll
        for (int r = 0; r < 8; ++r) {
            const int row = hi * 8 + r;
            Plds[w][row * 32 + lane15]      = (_Float16)p0[r];
            Plds[w][row * 32 + 16 + lane15] = (_Float16)p1[r];
        }
        __builtin_amdgcn_wave_barrier();
        union { v16h v; v4u u[2]; } pa;
        pa.u[0] = *(const v4u*)&Plds[w][lane15 * 32 + hi * 16];
        pa.u[1] = *(const v4u*)&Plds[w][lane15 * 32 + hi * 16 + 8];
        __builtin_amdgcn_wave_barrier();

        // ---- row sums: P @ ones -> every column holds the rowsum
        v8f cs = {};
        cs = __builtin_amdgcn_wmma_f32_16x16x32_f16(false, pa.v, false, bones,
                                                    (short)0, cs, false, false);

        // ---- O += P * V (tail clamps V address; P cols 16..31 are zero)
        const int kv = full ? (kk0 + hi * 16) : kk0;
        union { v16h v; v4u u[2]; } bv0, bv1;
        {
            const _Float16* vp0 = vbase + (size_t)lane15 * kNK + kv;
            const _Float16* vp1 = vbase + (size_t)(16 + lane15) * kNK + kv;
            bv0.u[0] = *(const v4u*)(vp0); bv0.u[1] = *(const v4u*)(vp0 + 8);
            bv1.u[0] = *(const v4u*)(vp1); bv1.u[1] = *(const v4u*)(vp1 + 8);
        }
        o0 = __builtin_amdgcn_wmma_f32_16x16x32_f16(false, pa.v, false, bv0.v,
                                                    (short)0, o0, false, false);
        o1 = __builtin_amdgcn_wmma_f32_16x16x32_f16(false, pa.v, false, bv1.v,
                                                    (short)0, o1, false, false);
        #pragma unroll
        for (int r = 0; r < 8; ++r)
            l_[r] = l_[r] * al[r] + cs[r];
    };

    // Branch-free main loop (round-robin chunks across the 4 waves), then tail.
    for (int c = w; c < kFullChunks; c += kSplit)
        chunk(c << 5, true);
    if (w == (kFullChunks % kSplit))
        chunk(kTail, false);

    // ---- publish partials, merge (exact LSE combine) by wave 0
    #pragma unroll
    for (int r = 0; r < 8; ++r) {
        const int row = hi * 8 + r;
        Olds[w][row][lane15]      = o0[r];
        Olds[w][row][16 + lane15] = o1[r];
        if (lane15 == 0) { Mlds[w][row] = m_[r]; Llds[w][row] = l_[r]; }
    }
    __syncthreads();
    if (w == 0) {
        #pragma unroll
        for (int r = 0; r < 8; ++r) {
            const int row = hi * 8 + r;
            const int q   = q0 + row;
            if (q >= kNQ) continue;
            float M = Mlds[0][row];
            #pragma unroll
            for (int ww = 1; ww < kSplit; ++ww) M = fmaxf(M, Mlds[ww][row]);
            float L = 0.f, a0 = 0.f, a1 = 0.f;
            #pragma unroll
            for (int ww = 0; ww < kSplit; ++ww) {
                const float e = __expf(Mlds[ww][row] - M);
                L  += Llds[ww][row] * e;
                a0 += Olds[ww][row][lane15] * e;
                a1 += Olds[ww][row][16 + lane15] * e;
            }
            const float inv = 1.f / L;
            float* xo = Xout + ((size_t)b * kNQ + q) * kC + h * kD;
            xo[lane15]      = a0 * inv;
            xo[16 + lane15] = a1 * inv;
        }
    }
}

// ---------------------------------------------------------------------------
extern "C" void kernel_launch(void* const* d_in, const int* in_sizes, int n_in,
                              void* d_out, int out_size, void* d_ws, size_t ws_size,
                              hipStream_t stream)
{
    const float* query = (const float*)d_in[0];
    const float* key   = (const float*)d_in[1];
    const float* value = (const float*)d_in[2];
    const float* rpe   = (const float*)d_in[3];
    const int*   kpm   = (const int*)  d_in[5];
    const float* qpos  = (const float*)d_in[6];
    const float* kpos  = (const float*)d_in[7];
    const float* Wq    = (const float*)d_in[8];
    const float* bq    = (const float*)d_in[9];
    const float* Wk    = (const float*)d_in[10];
    const float* bk    = (const float*)d_in[11];
    const float* Wv    = (const float*)d_in[12];
    const float* bv    = (const float*)d_in[13];
    const float* Wp    = (const float*)d_in[14];
    const float* bp    = (const float*)d_in[15];

    char* ws = (char*)d_ws;
    const size_t szQ = (size_t)kB * kH * kNQP * kD * sizeof(_Float16);
    const size_t szK = (size_t)kB * kH * kNK  * kD * sizeof(_Float16);
    const size_t szV = szK;
    const size_t offQ = 0;
    const size_t offK = offQ + ((szQ + 255) & ~(size_t)255);
    const size_t offV = offK + ((szK + 255) & ~(size_t)255);
    const size_t offX = offV + ((szV + 255) & ~(size_t)255);

    _Float16* Qh   = (_Float16*)(ws + offQ);
    _Float16* Kh   = (_Float16*)(ws + offK);
    _Float16* Vt   = (_Float16*)(ws + offV);
    float*    Xbuf = (float*)   (ws + offX);

    (void)hipMemsetAsync(Qh, 0, szQ, stream); // keep padded query rows finite

    const dim3 blk(32);
    const int MQ  = kB * kNQ;             // 1800 flat rows
    const int MK  = kB * kNK;             // 20000 flat rows
    const int mtQ = (MQ + 15) / 16;       // 113
    const int mtK = (MK + 15) / 16;       // 1250
    const int ntC = kC / 16;              // 16

    proj_kernel<0><<<dim3(mtQ, ntC), blk, 0, stream>>>(
        query, Wq, bq, qpos, MQ, kNQ, kNQP, kScale, Qh, nullptr);
    proj_kernel<0><<<dim3(mtK, ntC), blk, 0, stream>>>(
        key, Wk, bk, kpos, MK, kNK, kNK, 1.f, Kh, nullptr);
    proj_kernel<1><<<dim3(mtK, ntC), blk, 0, stream>>>(
        value, Wv, bv, nullptr, MK, kNK, kNK, 1.f, Vt, nullptr);

    attn_kernel<<<dim3(kNQP / 16, kH, kB), dim3(32 * kSplit), 0, stream>>>(
        Qh, Kh, Vt, rpe, kpm, Xbuf);

    proj_kernel<2><<<dim3(mtQ, ntC), blk, 0, stream>>>(
        Xbuf, Wp, bp, nullptr, MQ, kNQ, kNQ, 1.f, nullptr, (float*)d_out);
}
